// MymultiheadAttention_58969900974637
// MI455X (gfx1250) — compile-verified, matrix operand order checked by reading, not run on hardware
//
#include <hip/hip_runtime.h>

typedef __bf16 bf16;
typedef __attribute__((ext_vector_type(16))) __bf16 v16bf;
typedef __attribute__((ext_vector_type(8)))  __bf16 v8bf;
typedef __attribute__((ext_vector_type(4)))  __bf16 v4bf;
typedef __attribute__((ext_vector_type(2)))  __bf16 v2bf;
typedef __attribute__((ext_vector_type(8)))  float  v8f;

typedef unsigned int uint32x4 __attribute__((ext_vector_type(4)));
typedef int          int32x8  __attribute__((ext_vector_type(8)));
typedef int          int32x4  __attribute__((ext_vector_type(4)));

#if defined(__gfx1250__) && __has_builtin(__builtin_amdgcn_tensor_load_to_lds)
#define USE_TDM 1
#else
#define USE_TDM 0
#endif

#define WMMA_BF16(a, b, c) \
  __builtin_amdgcn_wmma_f32_16x16x32_bf16(false, (a), false, (b), (short)0, (c), false, false)

static __device__ __forceinline__ v16bf cat8(v8bf lo, v8bf hi) {
  return __builtin_shufflevector(lo, hi, 0, 1, 2, 3, 4, 5, 6, 7, 8, 9, 10, 11, 12, 13, 14, 15);
}

// ---------------------------------------------------------------------------
// C[M,N] = (A[M,K] @ W[N,K]^T + bias[N]) * scale
// A: InT (fp32 or bf16), W: fp32, out: OutT (bf16 or fp32).
// fp32 -> bf16 conversion fused into LDS staging (packed b64 stores).
// Block = 256 thr (8 waves), tile 128x128x32; wave grid 2x4; each wave
// computes a 4x2 grid of 16x16 WMMA tiles.
// ---------------------------------------------------------------------------
template <typename InT, typename OutT>
__global__ __launch_bounds__(256) void gemm_xwT_bias(
    const InT* __restrict__ A, const float* __restrict__ W,
    const float* __restrict__ bias, OutT* __restrict__ C,
    int M, int N, int K, float scale) {
  constexpr int BM = 128, BN = 128, BK = 32, LDA = BK + 8;  // 40 elems/row
  __shared__ __align__(16) bf16 As[BM * LDA];
  __shared__ __align__(16) bf16 Bs[BN * LDA];

  const int tid = threadIdx.x;
  const int lane = tid & 31;
  const int wave = tid >> 5;
  const int wm = wave >> 2;  // 0..1
  const int wn = wave & 3;   // 0..3
  const int mBase = blockIdx.y * BM;
  const int nBase = blockIdx.x * BN;
  const int lrow = lane & 15;
  const int lhalf = lane >> 4;

  const v8f vzero = {0.f, 0.f, 0.f, 0.f, 0.f, 0.f, 0.f, 0.f};
  v8f acc[4][2];
#pragma unroll
  for (int i = 0; i < 4; ++i)
#pragma unroll
    for (int j = 0; j < 2; ++j) acc[i][j] = vzero;

  for (int k0 = 0; k0 < K; k0 += BK) {
    __syncthreads();
    // Stage A-tile (128x32) and W-tile (128x32), converting to bf16.
    // 1024 4-elem chunks per tile, 4 per thread; packed 8-byte LDS stores.
#pragma unroll
    for (int c = 0; c < 4; ++c) {
      const int chunk = tid + c * 256;  // 0..1023
      const int row = chunk >> 3;       // 8 chunks per row
      const int col = (chunk & 7) * 4;
      const InT* ap = A + (size_t)(mBase + row) * K + k0 + col;
      v4bf ta;
#pragma unroll
      for (int e = 0; e < 4; ++e) ta[e] = (bf16)(float)ap[e];
      *(v4bf*)&As[row * LDA + col] = ta;
      const float* wp = W + (size_t)(nBase + row) * K + k0 + col;
      v4bf tb;
#pragma unroll
      for (int e = 0; e < 4; ++e) tb[e] = (bf16)wp[e];
      *(v4bf*)&Bs[row * LDA + col] = tb;
    }
    // Prefetch next k-tile while this one is consumed.
    if (k0 + BK < K) {
      const int row = tid >> 1;
      const int half = (tid & 1) * 16;
      __builtin_prefetch(A + (size_t)(mBase + row) * K + k0 + BK + half, 0, 0);
      __builtin_prefetch(W + (size_t)(nBase + row) * K + k0 + BK + half, 0, 0);
    }
    __syncthreads();

    v16bf a[4], b[2];
#pragma unroll
    for (int i = 0; i < 4; ++i) {
      const bf16* p = &As[(wm * 64 + i * 16 + lrow) * LDA + lhalf * 8];
      a[i] = cat8(*(const v8bf*)p, *(const v8bf*)(p + 16));
    }
#pragma unroll
    for (int j = 0; j < 2; ++j) {
      const bf16* p = &Bs[(wn * 32 + j * 16 + lrow) * LDA + lhalf * 8];
      b[j] = cat8(*(const v8bf*)p, *(const v8bf*)(p + 16));
    }
#pragma unroll
    for (int i = 0; i < 4; ++i)
#pragma unroll
      for (int j = 0; j < 2; ++j) acc[i][j] = WMMA_BF16(a[i], b[j], acc[i][j]);
  }

  // Epilogue. C/D layout: vgpr r -> M = r + 8*lhalf, N = lane&15.
#pragma unroll
  for (int i = 0; i < 4; ++i) {
#pragma unroll
    for (int j = 0; j < 2; ++j) {
      const int col = nBase + wn * 32 + j * 16 + lrow;
      const float bv = bias[col];
#pragma unroll
      for (int r = 0; r < 8; ++r) {
        const int row = mBase + wm * 64 + i * 16 + r + lhalf * 8;
        C[(size_t)row * N + col] = (OutT)((acc[i][j][r] + bv) * scale);
      }
    }
  }
}

// ---------------------------------------------------------------------------
// Flash attention over 32 groups of [2048, 64] bf16 (scale pre-folded into Q).
// Block = 256 thr (8 waves) handles 128 query rows of one group; each wave
// owns 16 rows. Computes S^T = K @ Q^T so the D-layout of scores is directly
// the B-fragment layout for O^T = V^T @ P^T (no LDS transpose of P).
// K blocks are staged by the Tensor Data Mover with DOUBLE BUFFERING:
// wave 0 issues block i+1's tensor_load_to_lds (pad_enable -> padded 72-elem
// LDS row stride) at the top of iteration i, then s_wait_tensorcnt(1) -- TDM
// ops of one wave complete in order, so cnt<=1 means the current buffer is
// ready while the next block's DMA overlaps the whole compute body.
// V is transposed into LDS with packed pair stores.
// ---------------------------------------------------------------------------
__global__ __launch_bounds__(256) void flash_attn(
    const bf16* __restrict__ Q, const bf16* __restrict__ K,
    const bf16* __restrict__ V, bf16* __restrict__ O) {
  constexpr int D = 64, LK = 2048, BKV = 128;
  constexpr int KS = D + 8;     // padded K row stride (72) - conflict-free b128
  constexpr int VTS = BKV + 8;  // padded V^T key stride (136)
  __shared__ __align__(16) bf16 Ks[2][BKV * KS];  // ping-pong [key][d]
  __shared__ __align__(16) bf16 Vt[D * VTS];      // [d][key]

  const int g = blockIdx.y;
  const int qB = blockIdx.x * 128;
  const int tid = threadIdx.x;
  const int lane = tid & 31, wave = tid >> 5;
  const int lrow = lane & 15, lhalf = lane >> 4;
  const int q0 = qB + wave * 16;

  const bf16* Qg = Q + (size_t)g * LK * D;
  const bf16* Kg = K + (size_t)g * LK * D;
  const bf16* Vg = V + (size_t)g * LK * D;
  bf16* Og = O + (size_t)g * LK * D;

#if USE_TDM
  auto issue_k_tdm = [&](int buf, int kb_) {
    // 2D TDM descriptor: tile 64x128 (elems x rows) of bf16, tensor row
    // stride 64; pad 4 DWORDs every 32 DWORDs -> LDS row stride 72 elems.
    const unsigned long long ga =
        (unsigned long long)(uintptr_t)(Kg + (size_t)kb_ * D);
    const unsigned int ldsb = (unsigned int)(uintptr_t)(&Ks[buf][0]);
    uint32x4 g0;
    g0[0] = 1u;  // count = 1 valid descriptor
    g0[1] = ldsb;
    g0[2] = (unsigned int)ga;
    g0[3] = ((unsigned int)(ga >> 32) & 0x01ffffffu) | (2u << 30);  // type=2
    int32x8 g1;
    g1[0] = (int)((1u << 16) |   // data_size = 2 bytes
                  (1u << 20) |   // pad_enable
                  (4u << 22) |   // pad_interval: every 32 DWORDs
                  (3u << 25));   // pad_amount: 4 DWORDs
    g1[1] = (int)(64u << 16);    // tensor_dim0 = 64
    g1[2] = (int)(2048u << 16);  // tensor_dim1 = 2048
    g1[3] = (int)(64u << 16);    // tile_dim0 = 64
    g1[4] = (int)128;            // tile_dim1 = 128
    g1[5] = (int)64;             // tensor_dim0_stride = 64
    g1[6] = 0;
    g1[7] = 0;
    int32x4 gz = {0, 0, 0, 0};
#if __clang_major__ >= 23
    int32x8 gz8 = {0, 0, 0, 0, 0, 0, 0, 0};
    __builtin_amdgcn_tensor_load_to_lds(g0, g1, gz, gz, gz8, 0);
#else
    __builtin_amdgcn_tensor_load_to_lds(g0, g1, gz, gz, 0);
#endif
  };
  if (wave == 0) issue_k_tdm(0, 0);  // preload first K block
#endif

  // Q as B-fragment (N = query col): row-major load, two d-steps of 32.
  v16bf qb[2];
  {
    const bf16* p = Qg + (size_t)(q0 + lrow) * D + lhalf * 8;
    qb[0] = cat8(*(const v8bf*)p, *(const v8bf*)(p + 16));
    qb[1] = cat8(*(const v8bf*)(p + 32), *(const v8bf*)(p + 48));
  }

  const v8f vzero = {0.f, 0.f, 0.f, 0.f, 0.f, 0.f, 0.f, 0.f};
  v8f o[4];  // O^T accum: 4 d-tiles x (16x16); lane q-col = lrow
#pragma unroll
  for (int t = 0; t < 4; ++t) o[t] = vzero;
  float m_i = -1e30f, l_i = 0.f;

  for (int it = 0; it < LK / BKV; ++it) {
    const int kb = it * BKV;
    const int cur = it & 1;
    const bool has_next = (kb + BKV) < LK;

    // Prefetch next V block into cache while this one computes.
    if (has_next) {
      __builtin_prefetch(Vg + (size_t)(kb + BKV) * D + (size_t)(tid & 127) * D, 0, 0);
    }
    __syncthreads();  // previous iteration's LDS fully consumed

#if USE_TDM
    if (wave == 0) {
      if (has_next) issue_k_tdm(cur ^ 1, kb + BKV);  // DMA next block now
    }
#else
    // Manual K staging: 1024 8-elem chunks, 4 per thread, b128 copies.
#pragma unroll
    for (int c = 0; c < 4; ++c) {
      const int chunk = tid + c * 256;
      const int row = chunk >> 3;
      const int col = (chunk & 7) * 8;
      *(v8bf*)&Ks[cur][row * KS + col] =
          *(const v8bf*)(Kg + (size_t)(kb + row) * D + col);
    }
#endif

    // Stage V^T [64][128]: thread owns a key pair x 16 dims, packed b32 stores.
    {
      const int kp2 = tid & 63;        // key pair index
      const int dq = (tid >> 6) * 16;  // d quarter
      const bf16* vp0 = Vg + (size_t)(kb + 2 * kp2) * D + dq;
      const v8bf c0a = *(const v8bf*)vp0;
      const v8bf c0b = *(const v8bf*)(vp0 + 8);
      const v8bf c1a = *(const v8bf*)(vp0 + D);
      const v8bf c1b = *(const v8bf*)(vp0 + D + 8);
#pragma unroll
      for (int e = 0; e < 8; ++e) {
        v2bf p0;
        p0[0] = c0a[e];
        p0[1] = c1a[e];
        *(v2bf*)&Vt[(dq + e) * VTS + 2 * kp2] = p0;
        v2bf p1;
        p1[0] = c0b[e];
        p1[1] = c1b[e];
        *(v2bf*)&Vt[(dq + 8 + e) * VTS + 2 * kp2] = p1;
      }
    }
#if USE_TDM
    if (wave == 0) {
      // In-order per-wave TDM completion: cnt<=1 -> current buffer is ready.
      if (has_next) __builtin_amdgcn_s_wait_tensorcnt(1);
      else          __builtin_amdgcn_s_wait_tensorcnt(0);
    }
#endif
    __syncthreads();

    // S^T = K @ Q^T : 8 key-tiles x 1 q-tile, two d-steps each.
    v8f s[8];
#pragma unroll
    for (int mt = 0; mt < 8; ++mt) {
      const bf16* kp = &Ks[cur][(mt * 16 + lrow) * KS + lhalf * 8];
      v16bf ka0 = cat8(*(const v8bf*)kp, *(const v8bf*)(kp + 16));
      v16bf ka1 = cat8(*(const v8bf*)(kp + 32), *(const v8bf*)(kp + 48));
      v8f accS = vzero;
      accS = WMMA_BF16(ka0, qb[0], accS);
      accS = WMMA_BF16(ka1, qb[1], accS);
      s[mt] = accS;
    }

    // Online softmax per query column (lanes n / n+16 share column lrow).
    float bm = -1e30f;
#pragma unroll
    for (int mt = 0; mt < 8; ++mt)
#pragma unroll
      for (int r = 0; r < 8; ++r) bm = fmaxf(bm, s[mt][r]);
    bm = fmaxf(bm, __shfl_xor(bm, 16, 32));
    const float m_new = fmaxf(m_i, bm);
    const float alpha = __expf(m_i - m_new);
    float ps = 0.f;
#pragma unroll
    for (int mt = 0; mt < 8; ++mt)
#pragma unroll
      for (int r = 0; r < 8; ++r) {
        const float p = __expf(s[mt][r] - m_new);
        s[mt][r] = p;
        ps += p;
      }
    ps += __shfl_xor(ps, 16, 32);
    l_i = l_i * alpha + ps;
    m_i = m_new;
#pragma unroll
    for (int t = 0; t < 4; ++t)
#pragma unroll
      for (int r = 0; r < 8; ++r) o[t][r] *= alpha;

    // Pack P^T into B-fragments: D-layout of s[] is already B-frag order.
    v16bf bp[4];
#pragma unroll
    for (int ks = 0; ks < 4; ++ks) {
      v16bf pk;
#pragma unroll
      for (int e = 0; e < 8; ++e) {
        pk[e] = (bf16)s[2 * ks][e];
        pk[8 + e] = (bf16)s[2 * ks + 1][e];
      }
      bp[ks] = pk;
    }

    // O^T += V^T @ P^T : 4 d-tiles x 4 key-steps.
#pragma unroll
    for (int dt = 0; dt < 4; ++dt) {
#pragma unroll
      for (int ks = 0; ks < 4; ++ks) {
        const bf16* vp2 = &Vt[(dt * 16 + lrow) * VTS + ks * 32 + lhalf * 8];
        v16bf va = cat8(*(const v8bf*)vp2, *(const v8bf*)(vp2 + 16));
        o[dt] = WMMA_BF16(va, bp[ks], o[dt]);
      }
    }
  }

  // Normalize and store O (bf16, row-major [q][d]).
  const float linv = 1.f / l_i;
#pragma unroll
  for (int dt = 0; dt < 4; ++dt)
#pragma unroll
    for (int r = 0; r < 8; ++r) {
      const int d = dt * 16 + r + lhalf * 8;
      Og[(size_t)(q0 + lrow) * D + d] = (bf16)(o[dt][r] * linv);
    }
}

// ---------------------------------------------------------------------------
extern "C" void kernel_launch(void* const* d_in, const int* in_sizes, int n_in,
                              void* d_out, int out_size, void* d_ws, size_t ws_size,
                              hipStream_t stream) {
  (void)in_sizes; (void)n_in; (void)out_size; (void)ws_size;
  const float* query = (const float*)d_in[0];
  const float* key_i = (const float*)d_in[1];
  const float* value = (const float*)d_in[2];
  const float* Wq = (const float*)d_in[3];
  const float* bq = (const float*)d_in[4];
  const float* Wk = (const float*)d_in[5];
  const float* bk = (const float*)d_in[6];
  const float* Wv = (const float*)d_in[7];
  const float* bv = (const float*)d_in[8];
  const float* Wo = (const float*)d_in[9];
  const float* bo = (const float*)d_in[10];
  float* out = (float*)d_out;

  const int M = 2 * 2048;  // B*L
  const int E = 1024;

  bf16* Qb = (bf16*)d_ws;  // [M, E] bf16
  bf16* Kb = Qb + (size_t)M * E;
  bf16* Vb = Kb + (size_t)M * E;
  bf16* Ob = Vb + (size_t)M * E;  // total 32 MB of workspace

  const dim3 gg(E / 128, M / 128);  // (8, 32)
  const float scaling = 0.125f;     // hd^-0.5, folded into Q projection

  gemm_xwT_bias<float, bf16><<<gg, 256, 0, stream>>>(query, Wq, bq, Qb, M, E, E, scaling);
  gemm_xwT_bias<float, bf16><<<gg, 256, 0, stream>>>(key_i, Wk, bk, Kb, M, E, E, 1.0f);
  gemm_xwT_bias<float, bf16><<<gg, 256, 0, stream>>>(value, Wv, bv, Vb, M, E, E, 1.0f);

  flash_attn<<<dim3(2048 / 128, 32), 256, 0, stream>>>(Qb, Kb, Vb, Ob);

  gemm_xwT_bias<bf16, float><<<gg, 256, 0, stream>>>(Ob, Wo, bo, out, M, E, E, 1.0f);
}
